// Graph_Sage_3556232921193
// MI455X (gfx1250) — matrix operand order repaired and checked by reading, not compiled, not run on hardware
//
#include <hip/hip_runtime.h>
#include <hip/hip_bf16.h>

typedef __attribute__((ext_vector_type(16))) __bf16 v16bf;
typedef __attribute__((ext_vector_type(8)))  float  v8f;

#define NN   2048
#define FH   12288   // H feature columns = C*2*B*L
#define GX   6144    // X feature columns = C*B*L

union TileV { uint4 q[2]; v16bf v; };

__device__ __forceinline__ unsigned short f2bf(float f) {
    unsigned u = __float_as_uint(f);
    u += 0x7FFFu + ((u >> 16) & 1u);           // round-to-nearest-even
    return (unsigned short)(u >> 16);
}
__device__ __forceinline__ float bf2f(unsigned short h) {
    return __uint_as_float(((unsigned)h) << 16);
}

// ---- prep: adjT[v][u] = bf16(adj[u][v]) ------------------------------------
__global__ __launch_bounds__(256) void k_adjT(const float* __restrict__ adj,
                                              unsigned short* __restrict__ adjT) {
    int t = blockIdx.x * 256 + threadIdx.x;          // NN*NN threads
    int v = t >> 11, u = t & (NN - 1);
    adjT[v * NN + u] = f2bf(adj[u * NN + v]);
}

// ---- prep: invdeg[v] = 1 / max(sum_u adj[u][v], 1) -------------------------
__global__ __launch_bounds__(256) void k_invdeg(const float* __restrict__ adj,
                                                float* __restrict__ invdeg) {
    int v = blockIdx.x * 256 + threadIdx.x;
    if (v >= NN) return;
    float s = 0.f;
    for (int u = 0; u < NN; ++u) s += adj[u * NN + v];
    invdeg[v] = 1.f / fmaxf(s, 1.f);
}

// ---- prep: H k=0 half = 4*x ------------------------------------------------
__global__ __launch_bounds__(256) void k_h0(const float* __restrict__ x,
                                            float* __restrict__ H) {
    int t = blockIdx.x * 256 + threadIdx.x;          // NN*GX threads
    int n = t / GX, g = t - n * GX;
    int c = g / 768, r = g - c * 768;                // r = b*24 + l
    int b = r / 24, l = r - b * 24;
    H[n * FH + c * 1536 + r] = 4.f * x[((b * 8 + c) * NN + n) * 24 + l];
}

// B-tile-ready packing: for K-block kb and column col, dword d (=0..15) holds
// bf16 rows K=kb*32+2d (lo16) and K=kb*32+2d+1 (hi16). Lane half l16 reads
// dwords [l16*8, l16*8+8) as two b128 loads.
__device__ __forceinline__ void pack32(const float a[32], unsigned int hi_d[16],
                                       unsigned int lo_d[16]) {
#pragma unroll
    for (int d = 0; d < 16; ++d) {
        unsigned short h0 = f2bf(a[2 * d]), h1 = f2bf(a[2 * d + 1]);
        unsigned short l0 = f2bf(a[2 * d] - bf2f(h0));
        unsigned short l1 = f2bf(a[2 * d + 1] - bf2f(h1));
        hi_d[d] = (unsigned)h0 | ((unsigned)h1 << 16);
        lo_d[d] = (unsigned)l0 | ((unsigned)l1 << 16);
    }
}

// ---- prep: blocked bf16 hi/lo of X (node dim = axis 2 of x) ----------------
__global__ __launch_bounds__(256) void k_xpk(const float* __restrict__ x,
                                             uint4* __restrict__ hi,
                                             uint4* __restrict__ lo) {
    int t = blockIdx.x * 256 + threadIdx.x;          // 64*GX threads
    int kb = t / GX, col = t - kb * GX;
    int c = col / 768, r = col - c * 768;
    int b = r / 24, l = r - b * 24;
    float a[32];
#pragma unroll
    for (int k = 0; k < 32; ++k)
        a[k] = x[((b * 8 + c) * NN + kb * 32 + k) * 24 + l];
    unsigned int hd[16], ld[16];
    pack32(a, hd, ld);
    int base = t * 4;                                // (kb*GX + col) * 4 uint4
#pragma unroll
    for (int q = 0; q < 4; ++q) {
        hi[base + q] = make_uint4(hd[4 * q], hd[4 * q + 1], hd[4 * q + 2], hd[4 * q + 3]);
        lo[base + q] = make_uint4(ld[4 * q], ld[4 * q + 1], ld[4 * q + 2], ld[4 * q + 3]);
    }
}

// ---- per-layer: blocked bf16 hi/lo of H ------------------------------------
__global__ __launch_bounds__(256) void k_hpk(const float* __restrict__ H,
                                             uint4* __restrict__ hi,
                                             uint4* __restrict__ lo) {
    int t = blockIdx.x * 256 + threadIdx.x;          // 64*FH threads
    int kb = t / FH, col = t - kb * FH;
    float a[32];
#pragma unroll
    for (int k = 0; k < 32; ++k)
        a[k] = H[(kb * 32 + k) * FH + col];
    unsigned int hd[16], ld[16];
    pack32(a, hd, ld);
    int base = t * 4;
#pragma unroll
    for (int q = 0; q < 4; ++q) {
        hi[base + q] = make_uint4(hd[4 * q], hd[4 * q + 1], hd[4 * q + 2], hd[4 * q + 3]);
        lo[base + q] = make_uint4(ld[4 * q], ld[4 * q + 1], ld[4 * q + 2], ld[4 * q + 3]);
    }
}

__device__ __forceinline__ v8f wmma_bf16(v16bf a, v16bf b, v8f c) {
    return __builtin_amdgcn_wmma_f32_16x16x32_bf16(false, a, false, b,
                                                   (short)0, c, false, false);
}

// ---- WMMA GEMM: OUT[v, col] = sum_u adjT[v][u] * (Bhi+Blo)[u, col] ---------
// Each wave: 4 M-tiles x 4 N-tiles (64x64 output). Out pitch fixed = FH.
// scatter==1 remaps column strip into H's interleaved k=1 feature columns.
__global__ __launch_bounds__(256)
void k_gemm(const unsigned short* __restrict__ adjT,
            const uint4* __restrict__ bhi,
            const uint4* __restrict__ blo,
            float* __restrict__ out,
            int cols, int scatter) {
    const int lane = threadIdx.x & 31;
    const int wave = blockIdx.x * 8 + (threadIdx.x >> 5);
    const int mg    = wave & 31;          // 32 groups of 64 rows
    const int strip = wave >> 5;          // 64-column strips
    const int m0 = mg * 64;
    const int n0 = strip * 64;
    const int lh  = lane & 15;
    const int l16 = lane >> 4;

    int ob = n0;
    if (scatter) ob = (n0 / 768) * 1536 + 768 + (n0 % 768);

    v8f zero = {0.f, 0.f, 0.f, 0.f, 0.f, 0.f, 0.f, 0.f};
    v8f acc[4][4];
#pragma unroll
    for (int i = 0; i < 4; ++i)
#pragma unroll
        for (int j = 0; j < 4; ++j) acc[i][j] = zero;

    const int aoff = l16 * 16;            // K sub-chunk per lane half

    for (int k0 = 0; k0 < NN; k0 += 32) {
        const int kb = k0 >> 5;
        TileV A[4];
#pragma unroll
        for (int i = 0; i < 4; ++i) {
            const unsigned short* ap = adjT + (m0 + i * 16 + lh) * NN + k0 + aoff;
            A[i].q[0] = *(const uint4*)(ap);
            A[i].q[1] = *(const uint4*)(ap + 8);
        }
#pragma unroll
        for (int j = 0; j < 4; ++j) {
            const int col = n0 + j * 16 + lh;
            const int bidx = (kb * cols + col) * 4 + l16 * 2;   // uint4 units
            TileV bh, bl;
            bh.q[0] = bhi[bidx]; bh.q[1] = bhi[bidx + 1];
            bl.q[0] = blo[bidx]; bl.q[1] = blo[bidx + 1];
#pragma unroll
            for (int i = 0; i < 4; ++i)
                acc[i][j] = wmma_bf16(A[i].v, bh.v, acc[i][j]);
#pragma unroll
            for (int i = 0; i < 4; ++i)
                acc[i][j] = wmma_bf16(A[i].v, bl.v, acc[i][j]);
        }
    }

#pragma unroll
    for (int i = 0; i < 4; ++i) {
        const int rbase = m0 + i * 16 + l16 * 8;     // C/D: M = r (+8 upper half)
#pragma unroll
        for (int j = 0; j < 4; ++j) {
            const int col = ob + j * 16 + lh;
#pragma unroll
            for (int r = 0; r < 8; ++r)
                out[(rbase + r) * FH + col] = acc[i][j][r];
        }
    }
}

// ---- per-layer apply: h = h*Ws^T + (agg/deg)*Wn^T + bias -------------------
__global__ __launch_bounds__(256)
void k_apply(float* __restrict__ H, const float* __restrict__ AGG,
             const float* __restrict__ Ws, const float* __restrict__ Wn,
             const float* __restrict__ bias, const float* __restrict__ invdeg,
             float* __restrict__ outp, int final_layer) {
    int t = blockIdx.x * 256 + threadIdx.x;          // NN*512 threads
    int n = t >> 9, vv = t & 511;                    // vv = (c*2+k)*32 + b
    int base = n * FH + vv * 24;
    float id = invdeg[n];
    float hv[24], av[24];
#pragma unroll
    for (int l = 0; l < 24; ++l) {
        hv[l] = H[base + l];
        av[l] = AGG[base + l] * id;
    }
    float o[24];
#pragma unroll 4
    for (int oo = 0; oo < 24; ++oo) {
        float s = bias[oo];
#pragma unroll
        for (int i = 0; i < 24; ++i)
            s += hv[i] * Ws[oo * 24 + i] + av[i] * Wn[oo * 24 + i];
        o[oo] = s;
    }
    if (!final_layer) {
#pragma unroll
        for (int oo = 0; oo < 24; ++oo) H[base + oo] = o[oo];
    } else {
        int ck = vv >> 5, b = vv & 31;               // out[b][c*2+k][n][l]
        int off = ((b * 16 + ck) * NN + n) * 24;
#pragma unroll
        for (int oo = 0; oo < 24; ++oo) outp[off + oo] = o[oo];
    }
}

extern "C" void kernel_launch(void* const* d_in, const int* in_sizes, int n_in,
                              void* d_out, int out_size, void* d_ws, size_t ws_size,
                              hipStream_t stream) {
    (void)in_sizes; (void)n_in; (void)out_size; (void)ws_size;
    const float* x    = (const float*)d_in[0];   // [B,C,N,L]
    const float* adj  = (const float*)d_in[1];   // [N,N]
    const float* Ws   = (const float*)d_in[2];   // [2,L,L]
    const float* Wn   = (const float*)d_in[3];   // [2,L,L]
    const float* bias = (const float*)d_in[4];   // [2,L]
    float* out = (float*)d_out;

    char* ws = (char*)d_ws;
    float*          H      = (float*)(ws);                      // 100,663,296 B
    float*          AGG    = (float*)(ws + 100663296u);         // 100,663,296 B
    uint4*          Hhi    = (uint4*)(ws + 201326592u);         //  50,331,648 B
    uint4*          Hlo    = (uint4*)(ws + 251658240u);         //  50,331,648 B
    unsigned short* adjT   = (unsigned short*)(ws + 301989888u);//   8,388,608 B
    float*          invdeg = (float*)(ws + 310378496u);         //       8 KiB

    // prep
    k_adjT  <<<(NN * NN) / 256, 256, 0, stream>>>(adj, adjT);
    k_invdeg<<<8, 256, 0, stream>>>(adj, invdeg);
    k_h0    <<<(NN * GX) / 256, 256, 0, stream>>>(x, H);
    k_xpk   <<<(64 * GX) / 256, 256, 0, stream>>>(x, Hhi, Hlo);

    // stage 1: H[:, k=1 cols] = adjT @ X   (scatter into interleaved layout)
    k_gemm<<<(32 * (GX / 64)) / 8, 256, 0, stream>>>(adjT, Hhi, Hlo, H, GX, 1);

    // two SAGE layers
    for (int i = 0; i < 2; ++i) {
        k_hpk <<<(64 * FH) / 256, 256, 0, stream>>>(H, Hhi, Hlo);
        k_gemm<<<(32 * (FH / 64)) / 8, 256, 0, stream>>>(adjT, Hhi, Hlo, AGG, FH, 0);
        k_apply<<<(NN * 512) / 256, 256, 0, stream>>>(H, AGG, Ws + i * 576, Wn + i * 576,
                                                      bias + i * 24, invdeg, out, i == 1);
    }
}
